// hhgnnConv_eu_adaptive_17927193494052
// MI455X (gfx1250) — matrix operand order repaired and checked by reading, not compiled, not run on hardware
//
#include <hip/hip_runtime.h>
#include <hip/hip_bf16.h>
#include <math.h>

#define HH   8      // heads
#define CC   16     // channels per head
#define HC   128    // H*C, also equals input dim IN
#define KDIM 128    // input feature dim

typedef __attribute__((ext_vector_type(16))) _Float16 v16h;
typedef __attribute__((ext_vector_type(8)))  float    v8f;

__device__ __forceinline__ void atomicMaxF(float* addr, float v) {
    // ordered-float trick; init value must be -inf (0xFF800000)
    if (v >= 0.0f) atomicMax((int*)addr, __float_as_int(v));
    else           atomicMin((unsigned int*)addr, (unsigned int)__float_as_int(v));
}

__global__ void fill_f32(float* __restrict__ p, float v, long n) {
    long i = (long)blockIdx.x * blockDim.x + threadIdx.x;
    long stride = (long)gridDim.x * blockDim.x;
    for (; i < n; i += stride) p[i] = v;
}

// ---------------------------------------------------------------------------
// Prepack W [4,128,128] f32 -> Wh f16 in WMMA B-fragment order:
//   Wh[((t*4 + kt)*2 + half)*128 + col][i] , i=0..15 contiguous,
//   holds W[t][kt*32 + half*8 + (i&7) + (i>>3)*16][col].
// A lane's whole B fragment becomes one aligned 32-byte load.
// ---------------------------------------------------------------------------
__global__ void pack_w_f16(const float* __restrict__ W, _Float16* __restrict__ Wh) {
    int idx = blockIdx.x * blockDim.x + threadIdx.x;      // 65536 total
    if (idx >= 4 * KDIM * HC) return;
    int i    = idx & 15;
    int col  = (idx >> 4) & 127;
    int half = (idx >> 11) & 1;
    int kt   = (idx >> 12) & 3;
    int t    = idx >> 14;
    int K    = kt * 32 + half * 8 + (i & 7) + (i >> 3) * 16;
    Wh[idx]  = (_Float16)W[((size_t)t * KDIM + K) * HC + col];
}

// ---------------------------------------------------------------------------
// Phase 1: typed GEMM  X0[n,:] = X[n,:] @ W[vtype[n]]  via WMMA f32<=f16
// block = 256 (8 waves); blockIdx.x = row tile (16 rows); wave id = col tile.
// Compute 4 accumulators (one per weight type), select per row at writeback.
// ---------------------------------------------------------------------------
__global__ void typed_gemm_wmma(const float* __restrict__ X,
                                const _Float16* __restrict__ Wh,  // packed fragments
                                const int*   __restrict__ vtype,
                                float*       __restrict__ X0,
                                int N) {
    const int lane = threadIdx.x & 31;
    const int wave = threadIdx.x >> 5;          // 0..7 -> column tile
    const int row0 = blockIdx.x * 16;
    const int m    = lane & 15;
    const int half = lane >> 4;                 // 0: lanes 0-15, 1: lanes 16-31
    const int col  = wave * 16 + m;
    const int rowA = (row0 + m < N) ? (row0 + m) : (N - 1);

    v8f acc[4] = {};

#pragma unroll
    for (int kt = 0; kt < 4; ++kt) {
        const int k = kt * 32;
        // A fragment: 16x32 f16. lanes 0-15: K = k+0..7 , k+16..23
        //             lanes 16-31: K = k+8..15, k+24..31   (ISA 7.12.2)
        const float4* xr = (const float4*)(X + (size_t)rowA * KDIM + k + half * 8);
        float4 xa = xr[0];          // K = base+0..3
        float4 xb = xr[1];          // K = base+4..7
        float4 xc = xr[4];          // K = base+16..19
        float4 xd = xr[5];          // K = base+20..23
        v16h a;
        a[0]  = (_Float16)xa.x; a[1]  = (_Float16)xa.y;
        a[2]  = (_Float16)xa.z; a[3]  = (_Float16)xa.w;
        a[4]  = (_Float16)xb.x; a[5]  = (_Float16)xb.y;
        a[6]  = (_Float16)xb.z; a[7]  = (_Float16)xb.w;
        a[8]  = (_Float16)xc.x; a[9]  = (_Float16)xc.y;
        a[10] = (_Float16)xc.z; a[11] = (_Float16)xc.w;
        a[12] = (_Float16)xd.x; a[13] = (_Float16)xd.y;
        a[14] = (_Float16)xd.z; a[15] = (_Float16)xd.w;

#pragma unroll
        for (int t = 0; t < 4; ++t) {
            // B fragment: one aligned 32B load from the packed layout
            const v16h b = *(const v16h*)(Wh +
                ((((size_t)t * 4 + kt) * 2 + half) * 128 + col) * 16);
            acc[t] = __builtin_amdgcn_wmma_f32_16x16x32_f16(
                false, a, false, b, (short)0, acc[t], false, false);
        }
    }

    // writeback: D vgpr r -> M = r + 8*half, N = col
#pragma unroll
    for (int r = 0; r < 8; ++r) {
        int row = row0 + r + half * 8;
        if (row < N) {
            int t = vtype[row];
            float val = (t == 0) ? acc[0][r] : (t == 1) ? acc[1][r]
                      : (t == 2) ? acc[2][r] : acc[3][r];
            X0[(size_t)row * HC + col] = val;
        }
    }
}

// ---------------------------------------------------------------------------
// Stage A pass 1: beta logits + segment max over edges.
// One wave per incidence; lane l covers floats [4l,4l+4) == head l>>2.
// ---------------------------------------------------------------------------
__global__ void att_edge_kernel(const float* __restrict__ X0,
                                const float* __restrict__ att_e,   // [4,128]
                                const int* __restrict__ vertex,
                                const int* __restrict__ edges,
                                const int* __restrict__ etype,
                                float* __restrict__ logits,        // [NNZ,8]
                                float* __restrict__ mbuf,          // [E,8] init -inf
                                int nnz) {
    int j    = blockIdx.x * (blockDim.x >> 5) + (threadIdx.x >> 5);
    int lane = threadIdx.x & 31;
    if (j >= nnz) return;
    int v = vertex[j], e = edges[j], t = etype[e];
    float4 x = ((const float4*)(X0    + (size_t)v * HC))[lane];
    float4 a = ((const float4*)(att_e + (size_t)t * HC))[lane];
    float p = x.x * a.x + x.y * a.y + x.z * a.z + x.w * a.w;
    p += __shfl_xor(p, 1, 32);
    p += __shfl_xor(p, 2, 32);
    p = (p > 0.0f) ? p : 0.2f * p;              // leaky relu
    if ((lane & 3) == 0) {
        int h = lane >> 2;
        logits[(size_t)j * HH + h] = p;
        atomicMaxF(&mbuf[(size_t)e * HH + h], p);
    }
}

// pass 2 (shared by both stages): ex = exp(logit - m[seg]); s[seg] += ex
__global__ void softmax_norm_kernel(const int* __restrict__ seg,
                                    float* __restrict__ logits,
                                    const float* __restrict__ mbuf,
                                    float* __restrict__ sbuf,
                                    long nElem) {
    long i = (long)blockIdx.x * blockDim.x + threadIdx.x;
    long stride = (long)gridDim.x * blockDim.x;
    for (; i < nElem; i += stride) {
        long j = i >> 3;
        int  h = (int)(i & 7);
        int  sg = seg[j];
        float ex = expf(logits[i] - mbuf[(size_t)sg * HH + h]);
        logits[i] = ex;
        atomicAdd(&sbuf[(size_t)sg * HH + h], ex);
    }
}

// Stage A pass 3: Xe[e] += beta * X0[vertex]
__global__ void scatter_edge_kernel(const float* __restrict__ X0,
                                    const float* __restrict__ logits,
                                    const float* __restrict__ sbuf,
                                    const int* __restrict__ vertex,
                                    const int* __restrict__ edges,
                                    float* __restrict__ Xe,
                                    int nnz) {
    int j    = blockIdx.x * (blockDim.x >> 5) + (threadIdx.x >> 5);
    int lane = threadIdx.x & 31;
    if (j >= nnz) return;
    int v = vertex[j], e = edges[j];
    int h = lane >> 2;
    float w = logits[(size_t)j * HH + h] / (sbuf[(size_t)e * HH + h] + 1e-16f);
    float4 x = ((const float4*)(X0 + (size_t)v * HC))[lane];
    float* dst = Xe + (size_t)e * HC + lane * 4;
    atomicAdd(dst + 0, x.x * w);
    atomicAdd(dst + 1, x.y * w);
    atomicAdd(dst + 2, x.z * w);
    atomicAdd(dst + 3, x.w * w);
}

// Stage B pass 1: alpha logits + segment max over vertices (relu(Xe) at read)
__global__ void att_vertex_kernel(const float* __restrict__ Xe,
                                  const float* __restrict__ att_v,   // [4,128]
                                  const int* __restrict__ vertex,
                                  const int* __restrict__ edges,
                                  const int* __restrict__ vtype,
                                  float* __restrict__ logits,
                                  float* __restrict__ mbuf,          // [N,8] init -inf
                                  int nnz) {
    int j    = blockIdx.x * (blockDim.x >> 5) + (threadIdx.x >> 5);
    int lane = threadIdx.x & 31;
    if (j >= nnz) return;
    int e = edges[j], v = vertex[j], t = vtype[v];
    float4 x = ((const float4*)(Xe + (size_t)e * HC))[lane];
    x.x = fmaxf(x.x, 0.0f); x.y = fmaxf(x.y, 0.0f);
    x.z = fmaxf(x.z, 0.0f); x.w = fmaxf(x.w, 0.0f);
    float4 a = ((const float4*)(att_v + (size_t)t * HC))[lane];
    float p = x.x * a.x + x.y * a.y + x.z * a.z + x.w * a.w;
    p += __shfl_xor(p, 1, 32);
    p += __shfl_xor(p, 2, 32);
    p = (p > 0.0f) ? p : 0.2f * p;
    if ((lane & 3) == 0) {
        int h = lane >> 2;
        logits[(size_t)j * HH + h] = p;
        atomicMaxF(&mbuf[(size_t)v * HH + h], p);
    }
}

// Stage B pass 3: out[v] += alpha * relu(Xe[edge])
__global__ void scatter_vertex_kernel(const float* __restrict__ Xe,
                                      const float* __restrict__ logits,
                                      const float* __restrict__ sbuf,
                                      const int* __restrict__ vertex,
                                      const int* __restrict__ edges,
                                      float* __restrict__ out,
                                      int nnz) {
    int j    = blockIdx.x * (blockDim.x >> 5) + (threadIdx.x >> 5);
    int lane = threadIdx.x & 31;
    if (j >= nnz) return;
    int e = edges[j], v = vertex[j];
    int h = lane >> 2;
    float w = logits[(size_t)j * HH + h] / (sbuf[(size_t)v * HH + h] + 1e-16f);
    float4 x = ((const float4*)(Xe + (size_t)e * HC))[lane];
    x.x = fmaxf(x.x, 0.0f); x.y = fmaxf(x.y, 0.0f);
    x.z = fmaxf(x.z, 0.0f); x.w = fmaxf(x.w, 0.0f);
    float* dst = out + (size_t)v * HC + lane * 4;
    atomicAdd(dst + 0, x.x * w);
    atomicAdd(dst + 1, x.y * w);
    atomicAdd(dst + 2, x.z * w);
    atomicAdd(dst + 3, x.w * w);
}

extern "C" void kernel_launch(void* const* d_in, const int* in_sizes, int n_in,
                              void* d_out, int out_size, void* d_ws, size_t ws_size,
                              hipStream_t stream) {
    const float* X     = (const float*)d_in[0];   // [N,128]
    const float* W     = (const float*)d_in[1];   // [4,128,128]
    const float* att_e = (const float*)d_in[2];   // [4,8,16]
    const float* att_v = (const float*)d_in[3];   // [4,8,16]
    const int* vertex  = (const int*)d_in[4];     // [NNZ]
    const int* edges   = (const int*)d_in[5];     // [NNZ]
    const int* vtype   = (const int*)d_in[6];     // [N]
    const int* etype   = (const int*)d_in[7];     // [E]
    float* out = (float*)d_out;

    const int N   = in_sizes[6];
    const int E   = in_sizes[7];
    const int NNZ = in_sizes[4];

    // workspace layout (in float units; every offset is a multiple of 8 floats
    // so all sub-buffers keep 32-byte alignment)
    size_t offX0  = 0;
    size_t offLog = offX0  + (size_t)N * HC;
    size_t offM   = offLog + (size_t)NNZ * HH;
    size_t offS   = offM   + (size_t)N * HH;
    size_t offXe  = offS   + (size_t)N * HH;
    size_t offWh  = offXe  + (size_t)E * HC;                    // f16 packed W
    size_t need   = (offWh + (size_t)(4 * KDIM * HC / 2)) * sizeof(float);
    if (ws_size < need) return;

    float*    X0     = (float*)d_ws + offX0;
    float*    logits = (float*)d_ws + offLog;
    float*    mbuf   = (float*)d_ws + offM;
    float*    sbuf   = (float*)d_ws + offS;
    float*    Xe     = (float*)d_ws + offXe;
    _Float16* Wh     = (_Float16*)((float*)d_ws + offWh);

    const float NEGINF = -__builtin_inff();

    // per-call re-init (atomics accumulate; harness does not re-poison)
    fill_f32<<<2048, 256, 0, stream>>>(out,  0.0f,  (long)N * HC);
    fill_f32<<<2048, 256, 0, stream>>>(Xe,   0.0f,  (long)E * HC);
    fill_f32<<<512,  256, 0, stream>>>(sbuf, 0.0f,  (long)N * HH);
    fill_f32<<<512,  256, 0, stream>>>(mbuf, NEGINF,(long)N * HH);

    // Phase 1: pack weights, then typed projection (WMMA)
    pack_w_f16<<<(4 * KDIM * HC + 255) / 256, 256, 0, stream>>>(W, Wh);
    typed_gemm_wmma<<<dim3((N + 15) / 16), 256, 0, stream>>>(X, Wh, vtype, X0, N);

    const int wavesPerBlk = 8;                     // 256 threads
    int gInc = (NNZ + wavesPerBlk - 1) / wavesPerBlk;
    long nElem = (long)NNZ * HH;
    int gExp = (int)((nElem + 255) / 256);

    // Stage A: vertex -> edge attention aggregation
    att_edge_kernel<<<gInc, 256, 0, stream>>>(X0, att_e, vertex, edges, etype,
                                              logits, mbuf, NNZ);
    softmax_norm_kernel<<<gExp, 256, 0, stream>>>(edges, logits, mbuf, sbuf, nElem);
    scatter_edge_kernel<<<gInc, 256, 0, stream>>>(X0, logits, sbuf, vertex, edges,
                                                  Xe, NNZ);

    // reset segment stats for stage B
    fill_f32<<<512, 256, 0, stream>>>(sbuf, 0.0f,   (long)N * HH);
    fill_f32<<<512, 256, 0, stream>>>(mbuf, NEGINF, (long)N * HH);

    // Stage B: edge -> vertex attention aggregation (relu folded into reads)
    att_vertex_kernel<<<gInc, 256, 0, stream>>>(Xe, att_v, vertex, edges, vtype,
                                                logits, mbuf, NNZ);
    softmax_norm_kernel<<<gExp, 256, 0, stream>>>(vertex, logits, mbuf, sbuf, nElem);
    scatter_vertex_kernel<<<gInc, 256, 0, stream>>>(Xe, logits, sbuf, vertex, edges,
                                                    out, NNZ);
}
